// FJSPActor_76338748720064
// MI455X (gfx1250) — matrix operand order repaired and checked by reading, not compiled, not run on hardware
//
#include <hip/hip_runtime.h>
#include <hip/hip_bf16.h>

typedef __attribute__((ext_vector_type(16))) __bf16 bf16x16;
typedef __attribute__((ext_vector_type(8)))  float  f32x8;

#define E_DIM 256
#define H_DIM 512
#define BS    32
#define NMA   64
#define NOPS  512
#define NJOBS 128
#define OUTW  8193   // 1 + 128*64

// ---------------------------------------------------------------------------
// Helpers
// ---------------------------------------------------------------------------
__device__ __forceinline__ unsigned short f2bf(float f) {
  unsigned int u = __builtin_bit_cast(unsigned int, f);
  unsigned int r = u + 0x7FFFu + ((u >> 16) & 1u);   // round-to-nearest-even
  return (unsigned short)(r >> 16);
}
__device__ __forceinline__ __bf16 bfbits(unsigned short u) {
  return __builtin_bit_cast(__bf16, u);
}

// Hardware V_TANH_F32 (CDNA5 trans op) instead of ocml polynomial tanh.
__device__ __forceinline__ float fast_tanh(float x) {
#if __has_builtin(__builtin_amdgcn_tanhf)
  return __builtin_amdgcn_tanhf(x);
#elif __has_builtin(__builtin_amdgcn_tanh_f32)
  return __builtin_amdgcn_tanh_f32(x);
#else
  float r;
  // trans-op hazard rule: 1 independent op or V_NOP before result use
  asm("v_tanh_f32 %0, %1\n\tv_nop" : "=v"(r) : "v"(x));
  return r;
#endif
}

// Async global->LDS copy (ASYNCcnt path). Builtin signature (from hipcc
// diagnostics): (v4i addrspace(1)*, v4i addrspace(3)*, imm offset, imm cpol).
#if __has_builtin(__builtin_amdgcn_global_load_async_to_lds_b128)
#define ASYNC_LDS 1
typedef __attribute__((__vector_size__(4 * sizeof(int)))) int v4i;
typedef __attribute__((address_space(1))) v4i* g_v4i_p;
typedef __attribute__((address_space(3))) v4i* l_v4i_p;
__device__ __forceinline__ void async_cp16(const unsigned short* g, unsigned short* l) {
  __builtin_amdgcn_global_load_async_to_lds_b128(
      (g_v4i_p)(unsigned long long)g,
      (l_v4i_p)(unsigned int)(unsigned long long)l, 0, 0);
}
__device__ __forceinline__ void async_wait0() {
#if __has_builtin(__builtin_amdgcn_s_wait_asynccnt)
  __builtin_amdgcn_s_wait_asynccnt(0);
#else
  asm volatile("s_wait_asynccnt 0x0" ::: "memory");
#endif
}
#endif

// ---------------------------------------------------------------------------
// Convert W1, W2 (f32 512x512 row-major) to bf16 workspace copies.
// ---------------------------------------------------------------------------
__global__ __launch_bounds__(256) void cvt_kernel(
    const float* __restrict__ W1, const float* __restrict__ W2,
    unsigned short* __restrict__ W1bf, unsigned short* __restrict__ W2bf) {
  const int i = blockIdx.x * 256 + threadIdx.x;
  if (i < H_DIM * H_DIM) {
    W1bf[i] = f2bf(W1[i]);
    W2bf[i] = f2bf(W2[i]);
  }
}

// ---------------------------------------------------------------------------
// Stage 1: HJ[b*128+j][h] = gather(ops_emb)@W1[:E] + b1   (4096 x 512, f32)
//          HM[b*64 +m][h] = ma_emb@W1[E:]                 (2048 x 512, f32)
// One wave = one 16-row strip, K=256 (8 bf16 WMMA chunks), N looped in 64s.
// Blocks 0..31 -> HJ strips, blocks 32..47 -> HM strips.
// ---------------------------------------------------------------------------
__global__ __launch_bounds__(256) void stage1_kernel(
    const float* __restrict__ ma_emb, const float* __restrict__ ops_emb,
    const int* __restrict__ next_op, const unsigned short* __restrict__ W1bf,
    const float* __restrict__ b1, float* __restrict__ HJ, float* __restrict__ HM) {
  __shared__ __attribute__((aligned(32))) unsigned short sB[E_DIM * 64]; // 32 KB
  const int tid  = threadIdx.x;
  const int lane = tid & 31;
  const int wave = tid >> 5;
  const int lo   = lane & 15;
  const int hi   = lane >> 4;

  const bool isHJ = (blockIdx.x < 32);
  const int strip = (isHJ ? blockIdx.x : (blockIdx.x - 32)) * 8 + wave;
  const int r0    = strip * 16;        // first output row of this strip
  const int rl    = r0 + lo;           // this lane's A-matrix row (M = lo)
  const int eoff  = isHJ ? 0 : E_DIM;  // which half of W1

  const float* src;
  if (isHJ) {
    const int b = rl >> 7, j = rl & 127;
    const int idx = next_op[b * NJOBS + j];
    src = ops_emb + ((long)b * NOPS + idx) * E_DIM;
  } else {
    src = ma_emb + (long)rl * E_DIM;   // rl == b*64+m
  }

  // A fragments (ISA 16-bit A 16x32 layout): lane M=lo, K halves split on hi.
  bf16x16 afr[8];
  const int kbase = hi ? 8 : 0;
#pragma unroll
  for (int kc = 0; kc < 8; ++kc) {
    const int kb = kc * 32 + kbase;
    float v[16];
    *(float4*)&v[0]  = *(const float4*)(src + kb);
    *(float4*)&v[4]  = *(const float4*)(src + kb + 4);
    *(float4*)&v[8]  = *(const float4*)(src + kb + 16);
    *(float4*)&v[12] = *(const float4*)(src + kb + 20);
#pragma unroll
    for (int t = 0; t < 16; ++t) afr[kc][t] = bfbits(f2bf(v[t]));
  }

  float* dst = isHJ ? HJ : HM;

  for (int ng = 0; ng < 8; ++ng) {
    __syncthreads();
    // cooperative tile load: W1bf rows eoff..eoff+255, cols ng*64..ng*64+63
#pragma unroll
    for (int it = 0; it < 8; ++it) {             // E_DIM*8 uint4 / 256 threads
      const int i = tid + it * 256;
      const int k = i >> 3, seg = i & 7;
      const unsigned short* g = W1bf + ((long)(eoff + k) * H_DIM + ng * 64 + seg * 8);
#ifdef ASYNC_LDS
      async_cp16(g, &sB[k * 64 + seg * 8]);
#else
      *(uint4*)&sB[k * 64 + seg * 8] = *(const uint4*)g;
#endif
    }
#ifdef ASYNC_LDS
    async_wait0();
#endif
    __syncthreads();

    f32x8 acc[4] = {};
#pragma unroll
    for (int kc = 0; kc < 8; ++kc) {
#pragma unroll
      for (int nt = 0; nt < 4; ++nt) {
        // B fragment (32x16): lane N = lo, elements over K in table order (hi splits K halves)
        bf16x16 bfr;
#pragma unroll
        for (int t = 0; t < 16; ++t)
          bfr[t] = bfbits(sB[(kc * 32 + (t < 8 ? 8 * hi + t : 16 + 8 * hi + (t - 8))) * 64 + nt * 16 + lo]);
        acc[nt] = __builtin_amdgcn_wmma_f32_16x16x32_bf16(
            false, afr[kc], false, bfr, (short)0, acc[nt], false, false);
      }
    }

    // C layout: VGPR r, lane -> M = r + 8*hi, N = lo
#pragma unroll
    for (int nt = 0; nt < 4; ++nt) {
      const int N = ng * 64 + nt * 16 + lo;
      const float bb = isHJ ? b1[N] : 0.f;
#pragma unroll
      for (int r = 0; r < 8; ++r) {
        const int row = r0 + r + 8 * hi;
        dst[(long)row * H_DIM + N] = acc[nt][r] + bb;
      }
    }
  }
}

// ---------------------------------------------------------------------------
// Stage 2 (fused): for each 16-row strip of (b,j,m) pairs:
//   A = tanh(hj + hm) as bf16 (16x512, kept fully in registers),
//   C = A @ W2bf (N looped 64 at a time, W2 tile staged in LDS),
//   out = sum_N tanh(C + b2[N]) * W3[N] + b3   (fused epilogue).
// 16384 strips, 8 waves/block -> 2048 blocks.
// ---------------------------------------------------------------------------
__global__ __launch_bounds__(256) void stage2_kernel(
    const float* __restrict__ HJ, const float* __restrict__ HM,
    const unsigned short* __restrict__ W2bf, const float* __restrict__ b2,
    const float* __restrict__ W3, const float* __restrict__ b3,
    float* __restrict__ out) {
  __shared__ __attribute__((aligned(32))) unsigned short sB[H_DIM * 64]; // 64 KB
  const int tid = threadIdx.x, lane = tid & 31, wave = tid >> 5;
  const int lo = lane & 15, hi = lane >> 4;

  const int s  = blockIdx.x * 8 + wave;  // strip id: (b, j, m-tile)
  const int b  = s >> 9;
  const int rm = s & 511;
  const int j  = rm >> 2;
  const int m0 = (rm & 3) * 16;

  const float* hjrow = HJ + (long)(b * NJOBS + j) * H_DIM;        // same all lanes
  const float* hmrow = HM + (long)(b * NMA + m0 + lo) * H_DIM;    // M = lo per lane

  // Build all 16 A fragments (K = 512) with the fused hardware tanh.
  bf16x16 afr[16];
  const int kbase = hi ? 8 : 0;
#pragma unroll
  for (int kc = 0; kc < 16; ++kc) {
    const int kb = kc * 32 + kbase;
    float vj[16], vm[16];
    *(float4*)&vj[0]  = *(const float4*)(hjrow + kb);
    *(float4*)&vj[4]  = *(const float4*)(hjrow + kb + 4);
    *(float4*)&vj[8]  = *(const float4*)(hjrow + kb + 16);
    *(float4*)&vj[12] = *(const float4*)(hjrow + kb + 20);
    *(float4*)&vm[0]  = *(const float4*)(hmrow + kb);
    *(float4*)&vm[4]  = *(const float4*)(hmrow + kb + 4);
    *(float4*)&vm[8]  = *(const float4*)(hmrow + kb + 16);
    *(float4*)&vm[12] = *(const float4*)(hmrow + kb + 20);
#pragma unroll
    for (int t = 0; t < 16; ++t)
      afr[kc][t] = bfbits(f2bf(fast_tanh(vj[t] + vm[t])));
  }

  float rowacc[8];
#pragma unroll
  for (int r = 0; r < 8; ++r) rowacc[r] = 0.f;

  for (int ng = 0; ng < 8; ++ng) {
    __syncthreads();
#pragma unroll
    for (int it = 0; it < 16; ++it) {            // H_DIM*8 uint4 / 256 threads
      const int i = tid + it * 256;
      const int k = i >> 3, seg = i & 7;
      const unsigned short* g = W2bf + ((long)k * H_DIM + ng * 64 + seg * 8);
#ifdef ASYNC_LDS
      async_cp16(g, &sB[k * 64 + seg * 8]);
#else
      *(uint4*)&sB[k * 64 + seg * 8] = *(const uint4*)g;
#endif
    }
#ifdef ASYNC_LDS
    async_wait0();
#endif
    __syncthreads();

    f32x8 acc[4] = {};
#pragma unroll
    for (int kc = 0; kc < 16; ++kc) {
#pragma unroll
      for (int nt = 0; nt < 4; ++nt) {
        bf16x16 bfr;
#pragma unroll
        for (int t = 0; t < 16; ++t)
          bfr[t] = bfbits(sB[(kc * 32 + (t < 8 ? 8 * hi + t : 16 + 8 * hi + (t - 8))) * 64 + nt * 16 + lo]);
        acc[nt] = __builtin_amdgcn_wmma_f32_16x16x32_bf16(
            false, afr[kc], false, bfr, (short)0, acc[nt], false, false);
      }
    }

    // fused epilogue: accumulate tanh(C + b2) * W3 per lane (lane owns N = lo)
#pragma unroll
    for (int nt = 0; nt < 4; ++nt) {
      const int N = ng * 64 + nt * 16 + lo;
      const float b2v = b2[N], w3v = W3[N];
#pragma unroll
      for (int r = 0; r < 8; ++r)
        rowacc[r] += fast_tanh(acc[nt][r] + b2v) * w3v;
    }
  }

  // reduce across the 16 lanes sharing each M
  const float b3v = b3[0];
#pragma unroll
  for (int r = 0; r < 8; ++r) {
#pragma unroll
    for (int off = 8; off; off >>= 1)
      rowacc[r] += __shfl_xor(rowacc[r], off, 32);
  }
  if (lo == 0) {
#pragma unroll
    for (int r = 0; r < 8; ++r) {
      const int m = m0 + r + 8 * hi;
      out[(long)b * OUTW + 1 + j * 64 + m] = rowacc[r] + b3v;
    }
  }
}

// ---------------------------------------------------------------------------
// No-op logit: dummy[512] through the MLP, broadcast to out[:, 0].
// ---------------------------------------------------------------------------
__global__ __launch_bounds__(512) void noop_kernel(
    const float* __restrict__ dummy, const float* __restrict__ W1,
    const float* __restrict__ b1, const float* __restrict__ W2,
    const float* __restrict__ b2, const float* __restrict__ W3,
    const float* __restrict__ b3, float* __restrict__ out) {
  __shared__ float sd[H_DIM];
  __shared__ float s1[H_DIM];
  __shared__ float red[16];
  const int t = threadIdx.x;
  sd[t] = dummy[t];
  __syncthreads();
  float a = b1[t];
  for (int e = 0; e < H_DIM; ++e) a += sd[e] * W1[e * H_DIM + t];
  s1[t] = fast_tanh(a);
  __syncthreads();
  float c = b2[t];
  for (int e = 0; e < H_DIM; ++e) c += s1[e] * W2[e * H_DIM + t];
  float p = fast_tanh(c) * W3[t];
  for (int off = 16; off; off >>= 1) p += __shfl_xor(p, off, 32);
  if ((t & 31) == 0) red[t >> 5] = p;
  __syncthreads();
  if (t == 0) {
    float sacc = 0.f;
    for (int i = 0; i < 16; ++i) sacc += red[i];
    red[0] = sacc + b3[0];
  }
  __syncthreads();
  if (t < 32) out[(long)t * OUTW] = red[0];
}

// ---------------------------------------------------------------------------
extern "C" void kernel_launch(void* const* d_in, const int* in_sizes, int n_in,
                              void* d_out, int out_size, void* d_ws, size_t ws_size,
                              hipStream_t stream) {
  const float* ma_emb  = (const float*)d_in[0];
  const float* ops_emb = (const float*)d_in[1];
  const int*   next_op = (const int*)d_in[2];
  const float* dummy   = (const float*)d_in[3];
  const float* W1      = (const float*)d_in[4];
  const float* b1      = (const float*)d_in[5];
  const float* W2      = (const float*)d_in[6];
  const float* b2      = (const float*)d_in[7];
  const float* W3      = (const float*)d_in[8];
  const float* b3      = (const float*)d_in[9];
  float* out = (float*)d_out;

  char* ws = (char*)d_ws;
  float* HJ = (float*)ws;                                   // 4096*512*4 = 8 MB
  float* HM = (float*)(ws + 8388608);                       // 2048*512*4 = 4 MB
  unsigned short* W1bf = (unsigned short*)(ws + 8388608 + 4194304);  // 512 KB
  unsigned short* W2bf = W1bf + H_DIM * H_DIM;                       // 512 KB

  cvt_kernel<<<1024, 256, 0, stream>>>(W1, W2, W1bf, W2bf);
  stage1_kernel<<<48, 256, 0, stream>>>(ma_emb, ops_emb, next_op, W1bf, b1, HJ, HM);
  stage2_kernel<<<2048, 256, 0, stream>>>(HJ, HM, W2bf, b2, W3, b3, out);
  noop_kernel<<<1, 512, 0, stream>>>(dummy, W1, b1, W2, b2, W3, b3, out);
}
